// SlotAttentionModule_3865470566660
// MI455X (gfx1250) — matrix operand order repaired and checked by reading, not compiled, not run on hardware
//
#include <hip/hip_runtime.h>
#include <math.h>

// ---------------- problem constants ----------------
#define Bn      32
#define Np      4096
#define CHn     256
#define DIMn    256
#define SLOTSn  11
#define HIDn    512
#define ITERSn  3
#define LN_EPS  1e-3f
#define EPS_ATTN 1e-8f
#define SCALEq  0.0625f   // 256^-0.5

// ---------------- workspace layout (bytes), needs ~140.2 MB ----------------
#define OFF_K    ((size_t)0)            // (B*N,256) bf16            64 MB
#define OFF_VT   ((size_t)67108864)     // (B,256,4096) bf16 d-major 64 MB
#define OFF_ATTN ((size_t)134217728)    // (B,16,4096) bf16           4 MB
#define OFF_Q    ((size_t)138412032)    // (B,16,256) bf16          256 KB
#define OFF_SLOT ((size_t)138674176)    // (B,16,256) f32           512 KB
#define OFF_UPD  ((size_t)139198464)    // (B,16,256) f32           512 KB
#define OFF_RS   ((size_t)139722752)    // (B,16) f32                 2 KB
#define OFF_WQ   ((size_t)139724800)    // (256,256) bf16           128 KB
#define OFF_WK2  ((size_t)139855872)    // (256,256) bf16           128 KB
#define OFF_WV2  ((size_t)139986944)    // (256,256) bf16           128 KB

// ---------------- WMMA plumbing ----------------
typedef __attribute__((ext_vector_type(16))) __bf16 v16bf;
typedef __attribute__((ext_vector_type(8)))  __bf16 v8bf;
typedef __attribute__((ext_vector_type(8)))  float  v8f;

union V16U { v16bf v; v8bf h[2]; };

static __device__ __forceinline__ v16bf ld_pair(const __bf16* p0, const __bf16* p1) {
  V16U u;
  u.h[0] = *(const v8bf*)p0;
  u.h[1] = *(const v8bf*)p1;
  return u.v;
}
static __device__ __forceinline__ v8f wmma_bf16(v16bf a, v16bf b, v8f c) {
  return __builtin_amdgcn_wmma_f32_16x16x32_bf16(false, a, false, b, (short)0, c, false, false);
}

// ---------------- async global->LDS staging (CDNA5 ASYNCcnt path) ----------------
#if __has_builtin(__builtin_amdgcn_global_load_async_to_lds_b128) && \
    __has_builtin(__builtin_amdgcn_s_wait_asynccnt)
#define HAVE_ASYNC_LDS 1
#else
#define HAVE_ASYNC_LDS 0
#endif

typedef int v4i_g __attribute__((vector_size(16)));   // matches builtin param pointee

static __device__ __forceinline__ void stage_w(const __bf16* __restrict__ src,
                                               __bf16* dst, int tid) {
#if HAVE_ASYNC_LDS
  for (int e = tid * 8; e < DIMn * CHn; e += 2048) {
    __builtin_amdgcn_global_load_async_to_lds_b128(
        (__attribute__((address_space(1))) v4i_g*)(src + e),
        (__attribute__((address_space(3))) v4i_g*)(dst + e), 0, 0);
  }
#else
  for (int e = tid * 8; e < DIMn * CHn; e += 2048)
    *(uint4*)(dst + e) = *(const uint4*)(src + e);
#endif
}
static __device__ __forceinline__ void wait_async_lds() {
#if HAVE_ASYNC_LDS
  __builtin_amdgcn_s_wait_asynccnt(0);
#endif
}

// ================= K0: slot init + Wq/Wk/Wv -> bf16 =================
__global__ void k0_init(const float* __restrict__ noise, const float* __restrict__ mu,
                        const float* __restrict__ ls, const float* __restrict__ wq_f,
                        const float* __restrict__ wk_f, const float* __restrict__ wv_f,
                        float* __restrict__ slots, __bf16* __restrict__ wqb,
                        __bf16* __restrict__ wkb, __bf16* __restrict__ wvb) {
  int idx = blockIdx.x * 256 + threadIdx.x;        // 0 .. 131071
  int row = idx >> 8, d = idx & 255;
  int b = row >> 4, i = row & 15;
  float v = 0.f;
  if (i < SLOTSn) v = mu[d] + expf(ls[d]) * noise[(size_t)(b * SLOTSn + i) * DIMn + d];
  slots[idx] = v;
  if (idx < DIMn * CHn) {
    wqb[idx] = (__bf16)wq_f[idx];
    wkb[idx] = (__bf16)wk_f[idx];
    wvb[idx] = (__bf16)wv_f[idx];
  }
}

// ================= K1: LN(inputs) + k / vT projection =================
// grid 512 (256 rows each), block 256 (8 waves), dyn LDS 262144
__global__ void k1_lnkv(const float* __restrict__ inp,
                        const __bf16* __restrict__ wkb, const __bf16* __restrict__ wvb,
                        const float* __restrict__ g_in, const float* __restrict__ be_in,
                        __bf16* __restrict__ kout, __bf16* __restrict__ vT) {
  extern __shared__ char smem[];
  __bf16* x_sh = (__bf16*)smem;            // 256*256 bf16 (128 KB)
  __bf16* w_sh = x_sh + 256 * CHn;         // 256*256 bf16 (128 KB)
  const int tid  = threadIdx.x;
  const int lane = tid & 31, wv = tid >> 5;
  const int hl = lane >> 4, ln16 = lane & 15;
  const int rowbase = blockIdx.x * 256;
  const int bb = rowbase >> 12;            // batch (256 | 4096)
  const int jloc0 = rowbase & 4095;

  // kick off Wk staging; DMA overlaps the LN phase below
  stage_w(wkb, w_sh, tid);

  float gv[8], bvv[8];
#pragma unroll
  for (int u = 0; u < 8; ++u) { gv[u] = g_in[lane * 8 + u]; bvv[u] = be_in[lane * 8 + u]; }

  // -------- LN: wave wv handles 32 rows --------
  for (int t = 0; t < 32; ++t) {
    int rl = wv * 32 + t;
    const float* rp = inp + (size_t)(rowbase + rl) * CHn + lane * 8;
    float4 a4 = ((const float4*)rp)[0];
    float4 b4 = ((const float4*)rp)[1];
    float x[8] = {a4.x, a4.y, a4.z, a4.w, b4.x, b4.y, b4.z, b4.w};
    float s = 0.f, ss = 0.f;
#pragma unroll
    for (int u = 0; u < 8; ++u) { s += x[u]; ss += x[u] * x[u]; }
#pragma unroll
    for (int off = 16; off > 0; off >>= 1) { s += __shfl_xor(s, off); ss += __shfl_xor(ss, off); }
    float m = s * (1.f / 256.f);
    float var = fmaxf(ss * (1.f / 256.f) - m * m, 0.f);
    float rstd = rsqrtf(var + LN_EPS);
    v8bf ov;
#pragma unroll
    for (int u = 0; u < 8; ++u) ov[u] = (__bf16)((x[u] - m) * rstd * gv[u] + bvv[u]);
    *(v8bf*)(x_sh + rl * CHn + lane * 8) = ov;
  }
  wait_async_lds();
  __syncthreads();

  // -------- k[j][d]: A = Wk (M=d), B = x (N=row j) --------
  // Wave owns dsub in {wv, wv+8} (outer), hoists that tile's 8 A-fragments (64 VGPRs),
  // and runs rsub row-pairs with two independent accumulator chains sharing A.
  for (int dsel = 0; dsel < 2; ++dsel) {
    const int dsub = wv + dsel * 8;
    v16bf A[8];
#pragma unroll
    for (int kk = 0; kk < 8; ++kk) {
      const __bf16* r0 = w_sh + (dsub * 16 + ln16) * CHn + kk * 32;
      A[kk] = ld_pair(r0 + hl * 16, r0 + hl * 16 + 8);
    }
    for (int r = 0; r < 8; ++r) {
      const __bf16* brow0 = x_sh + ((r)     * 16 + ln16) * CHn + hl * 16;
      const __bf16* brow1 = x_sh + ((r + 8) * 16 + ln16) * CHn + hl * 16;
      v8f c0 = {}, c1 = {};
#pragma unroll
      for (int kk = 0; kk < 8; ++kk) {
        v16bf b0 = ld_pair(brow0 + kk * 32, brow0 + kk * 32 + 8);
        v16bf b1 = ld_pair(brow1 + kk * 32, brow1 + kk * 32 + 8);
        c0 = wmma_bf16(A[kk], b0, c0);
        c1 = wmma_bf16(A[kk], b1, c1);
      }
      int j0 = rowbase + (r)     * 16 + ln16;
      int j1 = rowbase + (r + 8) * 16 + ln16;
      v8bf s0, s1;
#pragma unroll
      for (int t = 0; t < 8; ++t) { s0[t] = (__bf16)c0[t]; s1[t] = (__bf16)c1[t]; }
      *(v8bf*)(kout + (size_t)j0 * DIMn + dsub * 16 + hl * 8) = s0;
      *(v8bf*)(kout + (size_t)j1 * DIMn + dsub * 16 + hl * 8) = s1;
    }
  }
  __syncthreads();            // all waves done reading Wk from LDS
  stage_w(wvb, w_sh, tid);
  wait_async_lds();
  __syncthreads();

  // -------- vT[d][j]: A = x (M=row j), B = Wv^T (N=d) --------
  // Wave owns rsub in {wv, wv+8} (outer), hoists that tile's 8 A-fragments,
  // and runs dsub pairs with two accumulator chains sharing A.
  for (int rsel = 0; rsel < 2; ++rsel) {
    const int rsub = wv + rsel * 8;
    v16bf A[8];
#pragma unroll
    for (int kk = 0; kk < 8; ++kk) {
      const __bf16* r0 = x_sh + (rsub * 16 + ln16) * CHn + kk * 32;
      A[kk] = ld_pair(r0 + hl * 16, r0 + hl * 16 + 8);
    }
    const int j = jloc0 + rsub * 16 + hl * 8;
    for (int dd = 0; dd < 8; ++dd) {
      const __bf16* brow0 = w_sh + ((dd)     * 16 + ln16) * CHn + hl * 16;
      const __bf16* brow1 = w_sh + ((dd + 8) * 16 + ln16) * CHn + hl * 16;
      v8f c0 = {}, c1 = {};
#pragma unroll
      for (int kk = 0; kk < 8; ++kk) {
        v16bf b0 = ld_pair(brow0 + kk * 32, brow0 + kk * 32 + 8);
        v16bf b1 = ld_pair(brow1 + kk * 32, brow1 + kk * 32 + 8);
        c0 = wmma_bf16(A[kk], b0, c0);
        c1 = wmma_bf16(A[kk], b1, c1);
      }
      int d0 = (dd)     * 16 + ln16;
      int d1 = (dd + 8) * 16 + ln16;
      v8bf s0, s1;
#pragma unroll
      for (int t = 0; t < 8; ++t) { s0[t] = (__bf16)c0[t]; s1[t] = (__bf16)c1[t]; }
      *(v8bf*)(vT + ((size_t)(bb * DIMn + d0)) * Np + j) = s0;
      *(v8bf*)(vT + ((size_t)(bb * DIMn + d1)) * Np + j) = s1;
    }
  }
}

// ================= K2: slot-LN + q projection (+ zero rs/upd) =================
// grid 32 (batch), block 256
__global__ void k2_q(const float* __restrict__ slots, const float* __restrict__ g_sl,
                     const float* __restrict__ be_sl, const __bf16* __restrict__ wqb,
                     __bf16* __restrict__ q, float* __restrict__ rowsum,
                     float* __restrict__ upd) {
  __shared__ __align__(16) __bf16 sl_sh[16 * DIMn];
  const int b = blockIdx.x, tid = threadIdx.x;
  const int lane = tid & 31, wv = tid >> 5;
  const int hl = lane >> 4, ln16 = lane & 15;

  if (tid < 16) rowsum[b * 16 + tid] = 0.f;
  for (int e = tid; e < 16 * DIMn; e += 256) upd[(size_t)b * 16 * DIMn + e] = 0.f;

  float gv[8], bvv[8];
#pragma unroll
  for (int u = 0; u < 8; ++u) { gv[u] = g_sl[lane * 8 + u]; bvv[u] = be_sl[lane * 8 + u]; }
  for (int t = 0; t < 2; ++t) {
    int i = wv + t * 8;
    const float* rp = slots + (size_t)(b * 16 + i) * DIMn + lane * 8;
    float4 a4 = ((const float4*)rp)[0];
    float4 b4 = ((const float4*)rp)[1];
    float x[8] = {a4.x, a4.y, a4.z, a4.w, b4.x, b4.y, b4.z, b4.w};
    float s = 0.f, ss = 0.f;
#pragma unroll
    for (int u = 0; u < 8; ++u) { s += x[u]; ss += x[u] * x[u]; }
#pragma unroll
    for (int off = 16; off > 0; off >>= 1) { s += __shfl_xor(s, off); ss += __shfl_xor(ss, off); }
    float m = s * (1.f / 256.f);
    float var = fmaxf(ss * (1.f / 256.f) - m * m, 0.f);
    float rstd = rsqrtf(var + LN_EPS);
    v8bf ov;
#pragma unroll
    for (int u = 0; u < 8; ++u) ov[u] = (__bf16)((x[u] - m) * rstd * gv[u] + bvv[u]);
    *(v8bf*)(sl_sh + i * DIMn + lane * 8) = ov;
  }
  __syncthreads();
  // q[i][d]: A = ln(slots) (M=i, shared by tile pair), B = Wq^T (N=d)
  {
    const int dsub0 = 2 * wv, dsub1 = 2 * wv + 1;
    v8f c0 = {}, c1 = {};
    for (int kk = 0; kk < 8; ++kk) {
      int cb = kk * 32;
      const __bf16* ar = sl_sh + ln16 * DIMn + cb;
      v16bf a = ld_pair(ar + hl * 16, ar + hl * 16 + 8);
      const __bf16* b0 = wqb + (size_t)(dsub0 * 16 + ln16) * DIMn + cb + hl * 16;
      const __bf16* b1 = wqb + (size_t)(dsub1 * 16 + ln16) * DIMn + cb + hl * 16;
      c0 = wmma_bf16(a, ld_pair(b0, b0 + 8), c0);
      c1 = wmma_bf16(a, ld_pair(b1, b1 + 8), c1);
    }
#pragma unroll
    for (int v = 0; v < 8; ++v) {
      int i = v + hl * 8;
      q[(size_t)(b * 16 + i) * DIMn + dsub0 * 16 + ln16] = (__bf16)c0[v];
      q[(size_t)(b * 16 + i) * DIMn + dsub1 * 16 + ln16] = (__bf16)c1[v];
    }
  }
}

// ---- K3 epilogue: column softmax over 11 slots + store + rowsum ----
static __device__ __forceinline__ void softmax_store(v8f acc, int b, int j, int hl, int ln16,
                                                     __bf16* __restrict__ attn,
                                                     float* __restrict__ rowsum) {
  const int nvalid = hl ? 3 : 8;     // lanes<16 hold slots 0..7, lanes>=16 hold 8..15
  float dd[8];
#pragma unroll
  for (int v = 0; v < 8; ++v) dd[v] = acc[v] * SCALEq;
  float mx = -1e30f;
#pragma unroll
  for (int v = 0; v < 8; ++v) if (v < nvalid) mx = fmaxf(mx, dd[v]);
  mx = fmaxf(mx, __shfl_xor(mx, 16));
  float ev[8]; float se = 0.f;
#pragma unroll
  for (int v = 0; v < 8; ++v) { ev[v] = (v < nvalid) ? expf(dd[v] - mx) : 0.f; se += ev[v]; }
  se += __shfl_xor(se, 16);
  float inv = 1.f / se;
  float av[8];
#pragma unroll
  for (int v = 0; v < 8; ++v) av[v] = (v < nvalid) ? (ev[v] * inv + EPS_ATTN) : 0.f;
#pragma unroll
  for (int v = 0; v < 8; ++v) {
    int m = v + hl * 8;
    attn[((size_t)(b * 16 + m)) * Np + j] = (__bf16)av[v];
  }
#pragma unroll
  for (int v = 0; v < 8; ++v)
    for (int off = 1; off < 16; off <<= 1) av[v] += __shfl_xor(av[v], off, 16);
  if (ln16 == 0) {
#pragma unroll
    for (int v = 0; v < 8; ++v) {
      int m = v + hl * 8;
      if (m < SLOTSn) atomicAdd(&rowsum[b * 16 + m], av[v]);
    }
  }
}

// ================= K3: dots + slot-softmax + attn + rowsums =================
// grid (16 jtiles, 32 batches), block 256
__global__ void k3_attn(const __bf16* __restrict__ q, const __bf16* __restrict__ kmat,
                        __bf16* __restrict__ attn, float* __restrict__ rowsum) {
  __shared__ __align__(16) __bf16 q_sh[16 * DIMn];
  const int jt = blockIdx.x, b = blockIdx.y, tid = threadIdx.x;
  const int lane = tid & 31, wv = tid >> 5;
  const int hl = lane >> 4, ln16 = lane & 15;

  const uint4* qs = (const uint4*)(q + (size_t)b * 16 * DIMn);
  uint4* qd = (uint4*)q_sh;
  qd[tid] = qs[tid];
  qd[tid + 256] = qs[tid + 256];
  __syncthreads();

  const int j0 = jt * 256 + (2 * wv) * 16 + ln16;
  const int j1 = j0 + 16;
  v8f c0 = {}, c1 = {};
  for (int kk = 0; kk < 8; ++kk) {
    int cb = kk * 32;
    const __bf16* ar = q_sh + ln16 * DIMn + cb;   // A shared by the tile pair
    v16bf a = ld_pair(ar + hl * 16, ar + hl * 16 + 8);
    const __bf16* b0 = kmat + ((size_t)b * Np + j0) * DIMn + cb + hl * 16;
    const __bf16* b1 = kmat + ((size_t)b * Np + j1) * DIMn + cb + hl * 16;
    c0 = wmma_bf16(a, ld_pair(b0, b0 + 8), c0);
    c1 = wmma_bf16(a, ld_pair(b1, b1 + 8), c1);
  }
  softmax_store(c0, b, j0, hl, ln16, attn, rowsum);
  softmax_store(c1, b, j1, hl, ln16, attn, rowsum);
}

// ================= K4: updates_raw = attn @ v  (K = 4096) =================
// grid (8 jchunks, 32 batches), block 256; partial sums via f32 atomics
__global__ void k4_upd(const __bf16* __restrict__ attn, const __bf16* __restrict__ vT,
                       float* __restrict__ upd) {
  const int jc = blockIdx.x, b = blockIdx.y, tid = threadIdx.x;
  const int lane = tid & 31, wv = tid >> 5;
  const int hl = lane >> 4, ln16 = lane & 15;
  const int d0 = (2 * wv) * 16 + ln16;
  const int d1 = d0 + 16;
  v8f c0 = {}, c1 = {};
  for (int kk = 0; kk < 16; ++kk) {
    int jb = jc * 512 + kk * 32;
    const __bf16* ar = attn + ((size_t)(b * 16 + ln16)) * Np + jb;   // A shared
    v16bf a = ld_pair(ar + hl * 16, ar + hl * 16 + 8);
    const __bf16* b0 = vT + ((size_t)(b * DIMn + d0)) * Np + jb + hl * 16;
    const __bf16* b1 = vT + ((size_t)(b * DIMn + d1)) * Np + jb + hl * 16;
    c0 = wmma_bf16(a, ld_pair(b0, b0 + 8), c0);
    c1 = wmma_bf16(a, ld_pair(b1, b1 + 8), c1);
  }
#pragma unroll
  for (int v = 0; v < 8; ++v) {
    int i = v + hl * 8;
    atomicAdd(&upd[(size_t)(b * 16 + i) * DIMn + d0], c0[v]);
    atomicAdd(&upd[(size_t)(b * 16 + i) * DIMn + d1], c1[v]);
  }
}

// ================= K5: GRU + LN + FFN (f32, one block per batch) =================
// grid 32, block 512, dyn LDS 135168 B
__global__ void k5_gru_ffn(const float* __restrict__ upd, const float* __restrict__ rowsum,
                           const float* __restrict__ W_ih, const float* __restrict__ W_hh,
                           const float* __restrict__ b_ih, const float* __restrict__ b_hh,
                           const float* __restrict__ W1, const float* __restrict__ b1,
                           const float* __restrict__ W2, const float* __restrict__ b2,
                           const float* __restrict__ g_ff, const float* __restrict__ be_ff,
                           float* __restrict__ slots) {
  extern __shared__ float sm5[];
  float* u_sh  = sm5;                        // 11*256
  float* h_sh  = u_sh + SLOTSn * DIMn;       // 11*256
  float* gi_sh = h_sh + SLOTSn * DIMn;       // 3*11*256
  float* gh_sh = gi_sh + 3 * SLOTSn * DIMn;  // 3*11*256
  float* hn_sh = gh_sh + 3 * SLOTSn * DIMn;  // 11*256
  float* xn_sh = hn_sh + SLOTSn * DIMn;      // 11*256
  float* h1_sh = xn_sh + SLOTSn * DIMn;      // 11*512
  const int b = blockIdx.x, tid = threadIdx.x;

  for (int e = tid; e < SLOTSn * DIMn; e += 512) {
    int i = e >> 8, c = e & 255;
    int row = b * 16 + i;
    u_sh[e] = upd[(size_t)row * DIMn + c] / rowsum[row];
    h_sh[e] = slots[(size_t)row * DIMn + c];
  }
  __syncthreads();
  // ---- gates: threads <256 -> W_ih on updates, >=256 -> W_hh on slots ----
  {
    const int o = tid & 255;
    const bool is_ih = tid < 256;
    const float* W   = is_ih ? W_ih : W_hh;
    const float* src = is_ih ? u_sh : h_sh;
    const float* bia = is_ih ? b_ih : b_hh;
    float*       dst = is_ih ? gi_sh : gh_sh;
    float a0[SLOTSn], a1[SLOTSn], a2[SLOTSn];
#pragma unroll
    for (int i = 0; i < SLOTSn; ++i) { a0[i] = a1[i] = a2[i] = 0.f; }
    for (int c = 0; c < DIMn; c += 4) {
      float4 w0 = *(const float4*)(W + (size_t)o * DIMn + c);
      float4 w1 = *(const float4*)(W + (size_t)(o + 256) * DIMn + c);
      float4 w2 = *(const float4*)(W + (size_t)(o + 512) * DIMn + c);
#pragma unroll
      for (int i = 0; i < SLOTSn; ++i) {
        const float* sp = src + i * DIMn + c;
        float x0 = sp[0], x1 = sp[1], x2 = sp[2], x3 = sp[3];
        a0[i] += w0.x * x0 + w0.y * x1 + w0.z * x2 + w0.w * x3;
        a1[i] += w1.x * x0 + w1.y * x1 + w1.z * x2 + w1.w * x3;
        a2[i] += w2.x * x0 + w2.y * x1 + w2.z * x2 + w2.w * x3;
      }
    }
#pragma unroll
    for (int i = 0; i < SLOTSn; ++i) {
      dst[(0 * SLOTSn + i) * DIMn + o] = a0[i] + bia[o];
      dst[(1 * SLOTSn + i) * DIMn + o] = a1[i] + bia[256 + o];
      dst[(2 * SLOTSn + i) * DIMn + o] = a2[i] + bia[512 + o];
    }
  }
  __syncthreads();
  // ---- GRU elementwise ----
  for (int e = tid; e < SLOTSn * DIMn; e += 512) {
    int i = e >> 8, o = e & 255;
    float ir = gi_sh[(0 * SLOTSn + i) * DIMn + o];
    float iz = gi_sh[(1 * SLOTSn + i) * DIMn + o];
    float in_ = gi_sh[(2 * SLOTSn + i) * DIMn + o];
    float hr = gh_sh[(0 * SLOTSn + i) * DIMn + o];
    float hz = gh_sh[(1 * SLOTSn + i) * DIMn + o];
    float hn_ = gh_sh[(2 * SLOTSn + i) * DIMn + o];
    float r = 1.f / (1.f + expf(-(ir + hr)));
    float z = 1.f / (1.f + expf(-(iz + hz)));
    float n = tanhf(in_ + r * hn_);
    hn_sh[e] = (1.f - z) * n + z * h_sh[e];
  }
  __syncthreads();
  // ---- LN per slot row: wave w handles row w ----
  {
    const int wv = tid >> 5, lane = tid & 31;
    if (wv < SLOTSn) {
      float x[8]; float s = 0.f, ss = 0.f;
#pragma unroll
      for (int u = 0; u < 8; ++u) { float t = hn_sh[wv * DIMn + lane * 8 + u]; x[u] = t; s += t; ss += t * t; }
#pragma unroll
      for (int off = 16; off > 0; off >>= 1) { s += __shfl_xor(s, off); ss += __shfl_xor(ss, off); }
      float m = s * (1.f / 256.f);
      float var = fmaxf(ss * (1.f / 256.f) - m * m, 0.f);
      float rstd = rsqrtf(var + LN_EPS);
#pragma unroll
      for (int u = 0; u < 8; ++u) {
        int c = lane * 8 + u;
        xn_sh[wv * DIMn + c] = (x[u] - m) * rstd * g_ff[c] + be_ff[c];
      }
    }
  }
  __syncthreads();
  // ---- h1 = relu(xn @ W1^T + b1), 512 outputs <-> 512 threads ----
  {
    float acc[SLOTSn];
#pragma unroll
    for (int i = 0; i < SLOTSn; ++i) acc[i] = 0.f;
    for (int c = 0; c < DIMn; c += 4) {
      float4 wv4 = *(const float4*)(W1 + (size_t)tid * DIMn + c);
#pragma unroll
      for (int i = 0; i < SLOTSn; ++i) {
        const float* sp = xn_sh + i * DIMn + c;
        acc[i] += wv4.x * sp[0] + wv4.y * sp[1] + wv4.z * sp[2] + wv4.w * sp[3];
      }
    }
#pragma unroll
    for (int i = 0; i < SLOTSn; ++i) h1_sh[i * HIDn + tid] = fmaxf(acc[i] + b1[tid], 0.f);
  }
  __syncthreads();
  // ---- slots = gru_out + h1 @ W2^T + b2 (dot of 512, split over 2 half-threads) ----
  {
    const int o = tid & 255;
    const int cb = (tid >> 8) * 256;
    float acc[SLOTSn];
#pragma unroll
    for (int i = 0; i < SLOTSn; ++i) acc[i] = 0.f;
    for (int c = cb; c < cb + 256; c += 4) {
      float4 wv4 = *(const float4*)(W2 + (size_t)o * HIDn + c);
#pragma unroll
      for (int i = 0; i < SLOTSn; ++i) {
        const float* sp = h1_sh + i * HIDn + c;
        acc[i] += wv4.x * sp[0] + wv4.y * sp[1] + wv4.z * sp[2] + wv4.w * sp[3];
      }
    }
    if (tid >= 256) {
#pragma unroll
      for (int i = 0; i < SLOTSn; ++i) gi_sh[i * DIMn + o] = acc[i];  // scratch reuse
    }
    __syncthreads();
    if (tid < 256) {
#pragma unroll
      for (int i = 0; i < SLOTSn; ++i) {
        float val = hn_sh[i * DIMn + o] + acc[i] + gi_sh[i * DIMn + o] + b2[o];
        slots[(size_t)(b * 16 + i) * DIMn + o] = val;
      }
    }
  }
}

// ================= K6: strip padding rows into d_out =================
__global__ void k6_out(const float* __restrict__ slots, float* __restrict__ out) {
  const int blk = blockIdx.x, tid = threadIdx.x;  // 352 blocks
  const int b = blk / SLOTSn, i = blk % SLOTSn;
  out[(size_t)blk * DIMn + tid] = slots[(size_t)(b * 16 + i) * DIMn + tid];
}

// ================= host launcher =================
extern "C" void kernel_launch(void* const* d_in, const int* in_sizes, int n_in,
                              void* d_out, int out_size, void* d_ws, size_t ws_size,
                              hipStream_t stream) {
  const float* inputs = (const float*)d_in[0];
  const float* noise  = (const float*)d_in[1];
  const float* mu     = (const float*)d_in[2];
  const float* ls     = (const float*)d_in[3];
  const float* Wq     = (const float*)d_in[4];
  const float* Wk     = (const float*)d_in[5];
  const float* Wv     = (const float*)d_in[6];
  const float* W_ih   = (const float*)d_in[7];
  const float* W_hh   = (const float*)d_in[8];
  const float* b_ih   = (const float*)d_in[9];
  const float* b_hh   = (const float*)d_in[10];
  const float* W1     = (const float*)d_in[11];
  const float* b1     = (const float*)d_in[12];
  const float* W2     = (const float*)d_in[13];
  const float* b2     = (const float*)d_in[14];
  const float* g_in   = (const float*)d_in[15];
  const float* be_in  = (const float*)d_in[16];
  const float* g_sl   = (const float*)d_in[17];
  const float* be_sl  = (const float*)d_in[18];
  const float* g_ff   = (const float*)d_in[19];
  const float* be_ff  = (const float*)d_in[20];

  char* ws = (char*)d_ws;   // needs ~140.2 MB
  __bf16* kmat   = (__bf16*)(ws + OFF_K);
  __bf16* vT     = (__bf16*)(ws + OFF_VT);
  __bf16* attn   = (__bf16*)(ws + OFF_ATTN);
  __bf16* q      = (__bf16*)(ws + OFF_Q);
  float*  slots  = (float*)(ws + OFF_SLOT);
  float*  upd    = (float*)(ws + OFF_UPD);
  float*  rowsum = (float*)(ws + OFF_RS);
  __bf16* wqb    = (__bf16*)(ws + OFF_WQ);
  __bf16* wkb    = (__bf16*)(ws + OFF_WK2);
  __bf16* wvb    = (__bf16*)(ws + OFF_WV2);

  k0_init<<<512, 256, 0, stream>>>(noise, mu, ls, Wq, Wk, Wv, slots, wqb, wkb, wvb);
  k1_lnkv<<<512, 256, 262144, stream>>>(inputs, wkb, wvb, g_in, be_in, kmat, vT);
  for (int it = 0; it < ITERSn; ++it) {
    k2_q<<<32, 256, 0, stream>>>(slots, g_sl, be_sl, wqb, q, rowsum, upd);
    k3_attn<<<dim3(16, 32), 256, 0, stream>>>(q, kmat, attn, rowsum);
    k4_upd<<<dim3(8, 32), 256, 0, stream>>>(attn, vT, upd);
    k5_gru_ffn<<<32, 512, 135168, stream>>>(upd, rowsum, W_ih, W_hh, b_ih, b_hh,
                                            W1, b1, W2, b2, g_ff, be_ff, slots);
  }
  k6_out<<<352, 256, 0, stream>>>(slots, (float*)d_out);
}
